// TemporalAttentionLayer_69080253989479
// MI455X (gfx1250) — matrix operand order repaired and checked by reading, not compiled, not run on hardware
//
#include <hip/hip_runtime.h>
#include <hip/hip_bf16.h>

typedef __attribute__((ext_vector_type(16))) _Float16 v16h;
typedef __attribute__((ext_vector_type(8)))  _Float16 v8h;
typedef __attribute__((ext_vector_type(8)))  float    v8f;
typedef __attribute__((ext_vector_type(4))) unsigned int v4u;
typedef __attribute__((ext_vector_type(8)))  int      v8i_t;
typedef __attribute__((ext_vector_type(4)))  int      v4i_t;

#define NN 32768
#define TT 16
#define FF 256
#define HH 16
#define NODES_PER_WG 8

#if __has_builtin(__builtin_amdgcn_tensor_load_to_lds) && __has_builtin(__builtin_amdgcn_s_wait_tensorcnt)
#define TA_TDM 1
#else
#define TA_TDM 0
#endif

// ---------------------------------------------------------------------------
// Prep: pack Wq, Wk, Wv, lin_w^T into f16 WMMA B-fragment order.
// Layout: wp[((mat*16+nt)*8+kt)*32*16 + lane*16 + e]
//   value = W[k][n],  k = kt*32 + (e<8 ? (lane/16)*8 + e : 16 + (lane/16)*8 + e-8),  n = nt*16 + (lane%16)
// ---------------------------------------------------------------------------
__global__ void ta_pack_weights(const float* __restrict__ Wq,
                                const float* __restrict__ Wk,
                                const float* __restrict__ Wv,
                                const float* __restrict__ lw,
                                _Float16* __restrict__ wp)
{
    int idx  = blockIdx.x * blockDim.x + threadIdx.x;   // 262144 total
    int e    = idx & 15;
    int lane = (idx >> 4) & 31;
    int kt   = (idx >> 9) & 7;
    int nt   = (idx >> 12) & 15;
    int mat  = (idx >> 16) & 3;
    int k = kt * 32 + (e < 8 ? (lane >> 4) * 8 + e : 16 + (lane >> 4) * 8 + (e - 8));
    int n = nt * 16 + (lane & 15);
    float val;
    if      (mat == 0) val = Wq[k * FF + n];
    else if (mat == 1) val = Wk[k * FF + n];
    else if (mat == 2) val = Wv[k * FF + n];
    else               val = lw[n * FF + k];   // B[k][n] = lin_w[n][k] (pre-transposed)
    wp[idx] = (_Float16)val;
}

// ---------------------------------------------------------------------------
// Helpers
// ---------------------------------------------------------------------------
__device__ inline v8f zero8f() {
    v8f z;
#pragma unroll
    for (int i = 0; i < 8; ++i) z[i] = 0.0f;
    return z;
}

__device__ inline v8f wmma16(v16h a, v16h b, v8f c) {
    return __builtin_amdgcn_wmma_f32_16x16x32_f16(false, a, false, b,
                                                  (short)0, c, false, false);
}

__device__ inline v16h load_a16(const float* __restrict__ rowA,
                                const float* __restrict__ rowP,
                                int kt, int half)
{
    const int f0 = kt * 32 + half * 8;
    float4 p0 = *(const float4*)(rowA + f0);
    float4 p1 = *(const float4*)(rowA + f0 + 4);
    float4 p2 = *(const float4*)(rowA + f0 + 16);
    float4 p3 = *(const float4*)(rowA + f0 + 20);
    if (rowP) {
        float4 q0 = *(const float4*)(rowP + f0);
        float4 q1 = *(const float4*)(rowP + f0 + 4);
        float4 q2 = *(const float4*)(rowP + f0 + 16);
        float4 q3 = *(const float4*)(rowP + f0 + 20);
        p0.x += q0.x; p0.y += q0.y; p0.z += q0.z; p0.w += q0.w;
        p1.x += q1.x; p1.y += q1.y; p1.z += q1.z; p1.w += q1.w;
        p2.x += q2.x; p2.y += q2.y; p2.z += q2.z; p2.w += q2.w;
        p3.x += q3.x; p3.y += q3.y; p3.z += q3.z; p3.w += q3.w;
    }
    v16h a;
    a[0]  = (_Float16)p0.x; a[1]  = (_Float16)p0.y; a[2]  = (_Float16)p0.z; a[3]  = (_Float16)p0.w;
    a[4]  = (_Float16)p1.x; a[5]  = (_Float16)p1.y; a[6]  = (_Float16)p1.z; a[7]  = (_Float16)p1.w;
    a[8]  = (_Float16)p2.x; a[9]  = (_Float16)p2.y; a[10] = (_Float16)p2.z; a[11] = (_Float16)p2.w;
    a[12] = (_Float16)p3.x; a[13] = (_Float16)p3.y; a[14] = (_Float16)p3.z; a[15] = (_Float16)p3.w;
    return a;
}

#if TA_TDM
// Issue one TDM load of an 8 KB packed weight block into LDS (dynamic-LDS
// offset lds_off; dynamic LDS starts at 0, wbuf0=0 / wbuf1=8192).
// D# per CDNA5 ISA ch.8: 1-row tile, data_size=8B, tensor_dim0=tile_dim0=1024.
__device__ inline void tdm_issue(const _Float16* __restrict__ wp,
                                 int mat, int nt, unsigned lds_off)
{
    unsigned long long ga =
        (unsigned long long)(const void*)(wp + ((size_t)(mat * 16 + nt)) * 4096);
    v4u g0;
    g0[0] = 1u;                                        // count=1, user descriptor
    g0[1] = lds_off;                                   // lds_addr (bytes)
    g0[2] = (unsigned)(ga & 0xffffffffu);              // global_addr[31:0]
    g0[3] = (unsigned)((ga >> 32) & 0x01ffffffu)       // global_addr[56:32]
            | (2u << 30);                              // type = 2 ("image")
    v8i_t g1;
    g1[0] = (int)(3u << 16);                           // data_size = 8 bytes
    g1[1] = (int)(1024u << 16);                        // tensor_dim0[15:0] = 1024
    g1[2] = (int)(1u << 16);                           // tensor_dim1 = 1
    g1[3] = (int)(1024u << 16);                        // tile_dim0 = 1024
    g1[4] = 1;                                         // tile_dim1 = 1
    g1[5] = 1024;                                      // tensor_dim0_stride = 1024
    g1[6] = 0;
    g1[7] = 0;
    v4i_t z4; z4[0] = 0; z4[1] = 0; z4[2] = 0; z4[3] = 0;
#if defined(__clang_major__) && __clang_major__ >= 23
    v8i_t z8;
#pragma unroll
    for (int i = 0; i < 8; ++i) z8[i] = 0;
    __builtin_amdgcn_tensor_load_to_lds(g0, g1, z4, z4, z8, 0);
#else
    __builtin_amdgcn_tensor_load_to_lds(g0, g1, z4, z4, 0);
#endif
}
#endif

// Synchronous staging fallback (and used when TDM builtins are unavailable).
__device__ inline void stage_w(const _Float16* __restrict__ wp,
                               _Float16* __restrict__ wbuf, int mat, int nt, int tid)
{
    const float4* src = (const float4*)(wp + ((size_t)(mat * 16 + nt)) * 4096);
    float4* dst = (float4*)wbuf;
    dst[tid]       = src[tid];
    dst[tid + 256] = src[tid + 256];
}

// Open a GEMM phase: kick the first TDM prefetch.
__device__ inline void phase_open(const _Float16* __restrict__ wp, int mat, int wave)
{
#if TA_TDM
    if (wave == 0) tdm_issue(wp, mat, 0, 0u);
#endif
}

// Acquire weight tile nt for this phase; returns the LDS buffer to read.
// TDM path: wait for tile nt, one barrier, then prefetch tile nt+1 into the
// other buffer (last read in the previous iteration, so safe after this barrier).
__device__ inline const _Float16* phase_acquire(const _Float16* __restrict__ wp,
                                                _Float16* wbuf0, _Float16* wbuf1,
                                                int mat, int nt, int wave, int tid)
{
#if TA_TDM
    if (wave == 0) __builtin_amdgcn_s_wait_tensorcnt(0);
    __syncthreads();
    if (wave == 0 && nt + 1 < 16)
        tdm_issue(wp, mat, nt + 1, (unsigned)(((nt + 1) & 1) * 8192));
    return (nt & 1) ? wbuf1 : wbuf0;
#else
    __syncthreads();
    stage_w(wp, wbuf0, mat, nt, tid);
    __syncthreads();
    return wbuf0;
#endif
}

__device__ inline v8f gemm_tile(const v16h* a, const _Float16* __restrict__ wbuf, int lane)
{
    const v16h* wt = (const v16h*)wbuf;
    v8f acc = zero8f();
#pragma unroll
    for (int kt = 0; kt < 8; ++kt) acc = wmma16(a[kt], wt[kt * 32 + lane], acc);
    return acc;
}

// ---------------------------------------------------------------------------
// Main fused kernel: 8 waves/WG, one node per wave.
// ---------------------------------------------------------------------------
__global__ __launch_bounds__(256)
void ta_main(const float* __restrict__ in,   const float* __restrict__ EC,
             const float* __restrict__ pos,  const float* __restrict__ lin_b,
             const _Float16* __restrict__ wp,
             float* __restrict__ out, float* __restrict__ st)
{
    extern __shared__ float4 smem4[];
    _Float16* smem = reinterpret_cast<_Float16*>(smem4);

    const int tid  = threadIdx.x;
    const int lane = tid & 31;
    const int wave = tid >> 5;
    const int half = lane >> 4;
    const int l16  = lane & 15;
    const int node = blockIdx.x * NODES_PER_WG + wave;

    _Float16* wbuf0 = smem;                                  // 8 KB  (LDS off 0)
    _Float16* wbuf1 = smem + 4096;                           // 8 KB  (LDS off 8192)
    _Float16* qn   = smem + 8192 + wave * 4096;              // q   [16][256] f16
    _Float16* kn   = smem + 8192 +  8 * 4096 + wave * 4096;  // k   [16][256] f16
    _Float16* vTn  = smem + 8192 + 16 * 4096 + wave * 4096;  // v^T [256][16] f16
    _Float16* ctxn = smem + 8192 + 24 * 4096 + wave * 4096;  // ctx [16][256] f16
    _Float16* atn  = smem + 8192 + 32 * 4096 + wave * 256;   // attn scratch 16x16

    const float* inrow  = in  + ((size_t)node * TT + l16) * FF;
    const float* ecrow  = EC  + ((size_t)node * TT + l16) * FF;
    const float* posrow = pos + l16 * FF;

    v16h a[8];

    // ---- A fragments for temporal = inputs + pos_emb --------------------
#pragma unroll
    for (int kt = 0; kt < 8; ++kt) a[kt] = load_a16(inrow, posrow, kt, half);

    // ---- Q = temporal @ Wq  -> qn ---------------------------------------
    phase_open(wp, 0, wave);
#pragma unroll 1
    for (int nt = 0; nt < 16; ++nt) {
        const _Float16* wb = phase_acquire(wp, wbuf0, wbuf1, 0, nt, wave, tid);
        v8f acc = gemm_tile(a, wb, lane);
#pragma unroll
        for (int r = 0; r < 8; ++r)
            qn[(r + 8 * half) * FF + nt * 16 + l16] = (_Float16)acc[r];
    }

    // ---- V = temporal @ Wv  -> vTn (transposed, f-major) ----------------
    phase_open(wp, 2, wave);
#pragma unroll 1
    for (int nt = 0; nt < 16; ++nt) {
        const _Float16* wb = phase_acquire(wp, wbuf0, wbuf1, 2, nt, wave, tid);
        v8f acc = gemm_tile(a, wb, lane);
        v8h pk;
#pragma unroll
        for (int r = 0; r < 8; ++r) pk[r] = (_Float16)acc[r];
        *(v8h*)(vTn + ((nt * 16 + l16) * 16 + 8 * half)) = pk;   // vT[f][t]
    }

    // ---- K = EC @ Wk  -> kn ---------------------------------------------
#pragma unroll
    for (int kt = 0; kt < 8; ++kt) a[kt] = load_a16(ecrow, nullptr, kt, half);
    phase_open(wp, 1, wave);
#pragma unroll 1
    for (int nt = 0; nt < 16; ++nt) {
        const _Float16* wb = phase_acquire(wp, wbuf0, wbuf1, 1, nt, wave, tid);
        v8f acc = gemm_tile(a, wb, lane);
#pragma unroll
        for (int r = 0; r < 8; ++r)
            kn[(r + 8 * half) * FF + nt * 16 + l16] = (_Float16)acc[r];
    }

    // ---- Attention: per head 16x16 scores, softmax, ctx ------------------
    float* stn = st + (size_t)node * (TT * FF);
#pragma unroll 1
    for (int h = 0; h < HH; ++h) {
        // A fragment of q_h : M = l16 (query row), K = d (0..15 valid)
        v16h aq;
        {
#pragma unroll
            for (int e = 0; e < 16; ++e) aq[e] = (_Float16)0.0f;
            if (half == 0) aq = *(const v16h*)(qn + l16 * FF + h * 16);
        }
        // B fragment of k_h^T : N = key index, K = d (lanes 0-15 carry K=0..15)
        v16h bk;
#pragma unroll
        for (int e = 0; e < 16; ++e) bk[e] = (_Float16)0.0f;
        if (half == 0) bk = *(const v16h*)(kn + l16 * FF + h * 16);

        v8f s = wmma16(aq, bk, zero8f());

        // softmax over key axis (lanes within a 16-group), write ST + scratch
#pragma unroll
        for (int r = 0; r < 8; ++r) {
            float x = s[r] * 0.25f;                         // / sqrt(T)
            float m = x;
            m = fmaxf(m, __shfl_xor(m, 1, 32));
            m = fmaxf(m, __shfl_xor(m, 2, 32));
            m = fmaxf(m, __shfl_xor(m, 4, 32));
            m = fmaxf(m, __shfl_xor(m, 8, 32));
            float ex = __expf(x - m);
            float sm = ex;
            sm += __shfl_xor(sm, 1, 32);
            sm += __shfl_xor(sm, 2, 32);
            sm += __shfl_xor(sm, 4, 32);
            sm += __shfl_xor(sm, 8, 32);
            float p = ex / sm;
            int q = r + 8 * half;
            stn[q * FF + h * 16 + l16] = p;                 // ST[n][q][h*16+k]
            atn[q * 16 + l16] = (_Float16)p;                // transpose scratch
        }

        // ctx_h = attn_h @ v_h : A from atn (M=q row, K=k), B from vT (N=d, K=k)
        v16h aa;
        {
#pragma unroll
            for (int e = 0; e < 16; ++e) aa[e] = (_Float16)0.0f;
            if (half == 0) aa = *(const v16h*)(atn + l16 * 16);
        }
        v16h bv;
#pragma unroll
        for (int e = 0; e < 16; ++e) bv[e] = (_Float16)0.0f;
        if (half == 0) bv = *(const v16h*)(vTn + (h * 16 + l16) * 16);

        v8f c = wmma16(aa, bv, zero8f());
#pragma unroll
        for (int r = 0; r < 8; ++r)
            ctxn[(r + 8 * half) * FF + h * 16 + l16] = (_Float16)c[r];
    }

    // ---- FF: relu(ctx @ linT + b) + ctx + temporal, mean over T ----------
#pragma unroll
    for (int kt = 0; kt < 8; ++kt) {
        v8h c0 = *(const v8h*)(ctxn + l16 * FF + kt * 32 + half * 8);
        v8h c1 = *(const v8h*)(ctxn + l16 * FF + kt * 32 + 16 + half * 8);
#pragma unroll
        for (int e = 0; e < 8; ++e) { a[kt][e] = c0[e]; a[kt][e + 8] = c1[e]; }
    }
    phase_open(wp, 3, wave);
#pragma unroll 1
    for (int nt = 0; nt < 16; ++nt) {
        const _Float16* wb = phase_acquire(wp, wbuf0, wbuf1, 3, nt, wave, tid);
        v8f acc = gemm_tile(a, wb, lane);

        int f = nt * 16 + l16;
        float bias = lin_b[f];
        float partial = 0.0f;
#pragma unroll
        for (int r = 0; r < 8; ++r) {
            int t = r + 8 * half;
            float ctxv = (float)ctxn[t * FF + f];
            float ffv  = fmaxf(acc[r] + bias, 0.0f) + ctxv;
            float tmp  = in[((size_t)node * TT + t) * FF + f] + pos[t * FF + f];
            partial += ffv + tmp;
        }
        float tot = partial + __shfl_xor(partial, 16, 32);
        if (half == 0) out[(size_t)node * FF + f] = tot * 0.0625f;  // mean over T
    }
}

// ---------------------------------------------------------------------------
extern "C" void kernel_launch(void* const* d_in, const int* in_sizes, int n_in,
                              void* d_out, int out_size, void* d_ws, size_t ws_size,
                              hipStream_t stream)
{
    const float* in  = (const float*)d_in[0];
    const float* EC  = (const float*)d_in[1];
    const float* pos = (const float*)d_in[2];
    const float* Wq  = (const float*)d_in[3];
    const float* Wk  = (const float*)d_in[4];
    const float* Wv  = (const float*)d_in[5];
    const float* lw  = (const float*)d_in[6];
    const float* lb  = (const float*)d_in[7];

    float* out = (float*)d_out;                      // [N, F]
    float* st  = out + (size_t)NN * FF;              // [N, T, H*T]

    _Float16* wp = (_Float16*)d_ws;                  // 512 KB packed f16 weights

    ta_pack_weights<<<1024, 256, 0, stream>>>(Wq, Wk, Wv, lw, wp);

    const size_t lds_bytes = 282624;                 // 276 KB of the 320 KB WGP LDS
    hipFuncSetAttribute(reinterpret_cast<const void*>(ta_main),
                        hipFuncAttributeMaxDynamicSharedMemorySize,
                        (int)lds_bytes);
    ta_main<<<NN / NODES_PER_WG, 256, lds_bytes, stream>>>(in, EC, pos, lb, wp, out, st);
}
